// Encoder_13675175870584
// MI455X (gfx1250) — matrix operand order repaired and checked by reading, not compiled, not run on hardware
//
#include <hip/hip_runtime.h>
#include <cstdint>
#include <cstddef>

#define HEADS 4
#define T_LEN 91
#define A_NUM 64
#define G_RG 1400
#define G_TL 16
#define NEGV -1e10f

typedef __attribute__((ext_vector_type(16))) __bf16 v16bf;
typedef __attribute__((ext_vector_type(8)))  float  v8f;

union FragU { unsigned int u[8]; v16bf v; };

__device__ __forceinline__ unsigned short f2bf(float f) {
  unsigned int u = __float_as_uint(f);
  unsigned int r = ((u >> 16) & 1u) + 0x7FFFu;   // round-to-nearest-even
  return (unsigned short)((u + r) >> 16);
}
__device__ __forceinline__ unsigned pack2bf(float a, float b) {
  return (unsigned)f2bf(a) | ((unsigned)f2bf(b) << 16);
}
// gfx1250 async global->LDS 16-byte copy (raw bytes), tracked by ASYNCcnt.
__device__ __forceinline__ void async_b128(const void* g, void* lds) {
  unsigned l = (unsigned)(size_t)lds;          // low 32 bits = LDS byte offset
  asm volatile("global_load_async_to_lds_b128 %0, %1, off"
               :: "v"(l), "v"(g) : "memory");
}
__device__ __forceinline__ void async_wait0() {
  asm volatile("s_wait_asynccnt 0" ::: "memory");
}

// ---------------------------------------------------------------------------
// Pre-pack weights: Wt[n][k] = bf16(W[k][n])  (transposed so GEMM B-tiles are
// contiguous 16B chunks -> async-copyable straight into LDS)
// ---------------------------------------------------------------------------
__global__ void pack_wt(const float* __restrict__ W, unsigned short* __restrict__ Wt,
                        int K, int N) {
  size_t i = (size_t)blockIdx.x * blockDim.x + threadIdx.x;
  if (i >= (size_t)K * N) return;
  int n = (int)(i / K), k = (int)(i % K);
  Wt[i] = f2bf(W[(size_t)k * N + n]);
}

// ---------------------------------------------------------------------------
// GEMM: C[M,N] = epilogue(A[M,K] @ W[K,N]); W given as packed bf16 Wt[N][K].
// bf16 WMMA 16x16x32, f32 accum. Block tile 64x64, 4 wave32 waves.
// A: float4 loads + pack -> LDS b64 stores.  B: async b128 global->LDS.
// Epilogue staged through LDS for coalesced float4 stores.
// ---------------------------------------------------------------------------
__global__ __launch_bounds__(128) void gemm_bf16(
    const float* __restrict__ A, const unsigned short* __restrict__ Wt,
    const float* __restrict__ bias, const float* __restrict__ resid,
    const float* __restrict__ scale, float* __restrict__ C,
    int M, int N, int K, int relu)
{
  __shared__ alignas(16) unsigned short As[64][32];   // [m][k]
  __shared__ alignas(16) unsigned short Bs[64][32];   // [n][k]
  __shared__ alignas(16) float Cs[64][68];
  const int bm = blockIdx.x * 64, bn = blockIdx.y * 64;
  const int tid = threadIdx.x, wave = tid >> 5, lane = tid & 31;
  v8f acc[4];
#pragma unroll
  for (int i = 0; i < 4; ++i) acc[i] = (v8f){0,0,0,0,0,0,0,0};

  for (int k0 = 0; k0 < K; k0 += 32) {
    // B tile: 256 x 16B async chunks (2 per thread)
    for (int idx = tid; idx < 256; idx += 128) {
      int n = idx >> 2, ch = idx & 3;
      async_b128(Wt + (size_t)(bn + n) * K + k0 + ch * 8, &Bs[n][ch * 8]);
    }
    // A tile: 512 float4 chunks -> bf16-pack -> b64 LDS stores
    for (int idx = tid; idx < 512; idx += 128) {
      int r = idx >> 3, c4 = idx & 7;
      int gr = bm + r;
      float4 v = make_float4(0.f, 0.f, 0.f, 0.f);
      if (gr < M) v = *(const float4*)(A + (size_t)gr * K + k0 + c4 * 4);
      uint2 p; p.x = pack2bf(v.x, v.y); p.y = pack2bf(v.z, v.w);
      *(uint2*)&As[r][c4 * 4] = p;
    }
    async_wait0();
    __syncthreads();
    FragU fa;
    const int mrow = wave * 16 + (lane & 15);
    const int kba = (lane < 16) ? 0 : 16;
#pragma unroll
    for (int v = 0; v < 8; ++v) {
      int kk = kba + 2 * v;
      fa.u[v] = *(const unsigned int*)&As[mrow][kk];
    }
    const int kbb = (lane < 16) ? 0 : 16;
#pragma unroll
    for (int nb = 0; nb < 4; ++nb) {
      FragU fb;
      const int ncol = nb * 16 + (lane & 15);
#pragma unroll
      for (int v = 0; v < 8; ++v)
        fb.u[v] = *(const unsigned int*)&Bs[ncol][kbb + 2 * v];
      acc[nb] = __builtin_amdgcn_wmma_f32_16x16x32_bf16(
          false, fa.v, false, fb.v, (short)0, acc[nb], false, false);
    }
    __syncthreads();
  }
  // spill accumulators to LDS (C/D layout: lane= n, vgpr idx = m)
  {
    const int ml = wave * 16 + ((lane >= 16) ? 8 : 0);
    const int nl = lane & 15;
#pragma unroll
    for (int nb = 0; nb < 4; ++nb)
#pragma unroll
      for (int r = 0; r < 8; ++r)
        Cs[ml + r][nb * 16 + nl] = acc[nb][r];
  }
  __syncthreads();
  // coalesced epilogue stores
  for (int i = tid; i < 64 * 16; i += 128) {
    int r = i >> 4, c4 = i & 15;
    int m = bm + r;
    if (m >= M) continue;
    int n0 = bn + c4 * 4;
    float4 y = *(float4*)&Cs[r][c4 * 4];
    if (bias) {
      float4 bv = *(const float4*)(bias + n0);
      y.x += bv.x; y.y += bv.y; y.z += bv.z; y.w += bv.w;
    }
    if (relu) {
      y.x = fmaxf(y.x, 0.f); y.y = fmaxf(y.y, 0.f);
      y.z = fmaxf(y.z, 0.f); y.w = fmaxf(y.w, 0.f);
    }
    if (scale) {
      float4 sc = *(const float4*)(scale + (n0 & 63));
      y.x *= sc.x; y.y *= sc.y; y.z *= sc.z; y.w *= sc.w;
    }
    if (resid) {
      float4 rv = *(const float4*)(resid + (size_t)m * N + n0);
      y.x += rv.x; y.y += rv.y; y.z += rv.z; y.w += rv.w;
    }
    *(float4*)(C + (size_t)m * N + n0) = y;
  }
}

// ---------------------------------------------------------------------------
// Batched attention scores: E[b,i,j] = 0.5 * sum_d Q_b[i,d] * K_b[j,d]
// mode 0: batches=(a,h), rows=t | mode 1/2: batches=(t,h), rows=a / keys=g
// ---------------------------------------------------------------------------
__global__ __launch_bounds__(128) void attn_qk(
    const float* __restrict__ Q, const float* __restrict__ Kp,
    float* __restrict__ E, int mode, int M, int N, int Npad, int T)
{
  __shared__ alignas(16) unsigned short As[64][32];
  __shared__ alignas(16) unsigned short Bs[64][32];
  __shared__ alignas(16) float Cs[64][68];
  const int b = blockIdx.z, h = b & 3, bt = b >> 2;
  size_t base, str;
  if (mode == 0) { base = (size_t)bt * T * 256 + h * 64; str = 256; }
  else           { base = (size_t)bt * 256 + h * 64;     str = (size_t)T * 256; }
  const int bm0 = blockIdx.x * 64, bn0 = blockIdx.y * 64;
  const int tid = threadIdx.x, wave = tid >> 5, lane = tid & 31;
  v8f acc[4];
#pragma unroll
  for (int i = 0; i < 4; ++i) acc[i] = (v8f){0,0,0,0,0,0,0,0};

  for (int k0 = 0; k0 < 64; k0 += 32) {
    for (int idx = tid; idx < 512; idx += 128) {
      int r = idx >> 3, c4 = idx & 7;
      int gi = bm0 + r;
      float4 v = make_float4(0.f, 0.f, 0.f, 0.f);
      if (gi < M) v = *(const float4*)(Q + base + (size_t)gi * str + k0 + c4 * 4);
      uint2 p; p.x = pack2bf(v.x, v.y); p.y = pack2bf(v.z, v.w);
      *(uint2*)&As[r][c4 * 4] = p;
    }
    for (int idx = tid; idx < 512; idx += 128) {
      int r = idx >> 3, c4 = idx & 7;
      int gn = bn0 + r;
      float4 v = make_float4(0.f, 0.f, 0.f, 0.f);
      if (gn < N) v = *(const float4*)(Kp + base + (size_t)gn * str + k0 + c4 * 4);
      uint2 p; p.x = pack2bf(v.x, v.y); p.y = pack2bf(v.z, v.w);
      *(uint2*)&Bs[r][c4 * 4] = p;
    }
    __syncthreads();
    FragU fa;
    const int mrow = wave * 16 + (lane & 15);
    const int kba = (lane < 16) ? 0 : 16;
#pragma unroll
    for (int v = 0; v < 8; ++v) {
      int kk = kba + 2 * v;
      fa.u[v] = *(const unsigned int*)&As[mrow][kk];
    }
    const int kbb = (lane < 16) ? 0 : 16;
#pragma unroll
    for (int nb = 0; nb < 4; ++nb) {
      FragU fb;
      const int ncol = nb * 16 + (lane & 15);
#pragma unroll
      for (int v = 0; v < 8; ++v)
        fb.u[v] = *(const unsigned int*)&Bs[ncol][kbb + 2 * v];
      acc[nb] = __builtin_amdgcn_wmma_f32_16x16x32_bf16(
          false, fa.v, false, fb.v, (short)0, acc[nb], false, false);
    }
    __syncthreads();
  }
  {
    const int ml = wave * 16 + ((lane >= 16) ? 8 : 0);
    const int nl = lane & 15;
#pragma unroll
    for (int nb = 0; nb < 4; ++nb)
#pragma unroll
      for (int r = 0; r < 8; ++r)
        Cs[ml + r][nb * 16 + nl] = acc[nb][r];
  }
  __syncthreads();
  for (int i = tid; i < 64 * 16; i += 128) {
    int r = i >> 4, c4 = i & 15;
    int m = bm0 + r;
    if (m >= M) continue;                 // padding cols of E are ok to write
    float4 y = *(float4*)&Cs[r][c4 * 4];
    y.x *= 0.5f; y.y *= 0.5f; y.z *= 0.5f; y.w *= 0.5f;   // / sqrt(H=4)
    *(float4*)&E[((size_t)b * M + m) * Npad + bn0 + c4 * 4] = y;
  }
}

// ---------------------------------------------------------------------------
// Batched P @ V : Y_b[i,d] = sum_s P[b,i,s] * V_b[s,d]  (d = 0..63)
// ---------------------------------------------------------------------------
__global__ __launch_bounds__(128) void attn_av(
    const float* __restrict__ P, const float* __restrict__ V,
    float* __restrict__ Y, int mode, int M, int Nk, int Npad, int T)
{
  __shared__ alignas(16) unsigned short As[64][32];
  __shared__ alignas(16) unsigned short Bs[64][32];
  __shared__ alignas(16) float Cs[64][68];
  const int b = blockIdx.z, h = b & 3, bt = b >> 2;
  size_t base, str;
  if (mode == 0) { base = (size_t)bt * T * 256 + h * 64; str = 256; }
  else           { base = (size_t)bt * 256 + h * 64;     str = (size_t)T * 256; }
  const int bm0 = blockIdx.x * 64;
  const int tid = threadIdx.x, wave = tid >> 5, lane = tid & 31;
  const int Kpad = (Nk + 31) & ~31;
  v8f acc[4];
#pragma unroll
  for (int i = 0; i < 4; ++i) acc[i] = (v8f){0,0,0,0,0,0,0,0};

  for (int k0 = 0; k0 < Kpad; k0 += 32) {
    for (int idx = tid; idx < 512; idx += 128) {   // P rows, k-contiguous
      int r = idx >> 3, c4 = idx & 7;
      int gi = bm0 + r;
      float4 v = make_float4(0.f, 0.f, 0.f, 0.f);
      if (gi < M) v = *(const float4*)(P + ((size_t)b * M + gi) * Npad + k0 + c4 * 4);
      uint2 p; p.x = pack2bf(v.x, v.y); p.y = pack2bf(v.z, v.w);
      *(uint2*)&As[r][c4 * 4] = p;
    }
    for (int idx = tid; idx < 512; idx += 128) {   // V: vector global, scatter LDS
      int kk = idx >> 4, c4 = idx & 15;
      float4 v = make_float4(0.f, 0.f, 0.f, 0.f);
      if (k0 + kk < Nk) v = *(const float4*)(V + base + (size_t)(k0 + kk) * str + c4 * 4);
      Bs[c4 * 4 + 0][kk] = f2bf(v.x);
      Bs[c4 * 4 + 1][kk] = f2bf(v.y);
      Bs[c4 * 4 + 2][kk] = f2bf(v.z);
      Bs[c4 * 4 + 3][kk] = f2bf(v.w);
    }
    __syncthreads();
    FragU fa;
    const int mrow = wave * 16 + (lane & 15);
    const int kba = (lane < 16) ? 0 : 16;
#pragma unroll
    for (int v = 0; v < 8; ++v) {
      int kk = kba + 2 * v;
      fa.u[v] = *(const unsigned int*)&As[mrow][kk];
    }
    const int kbb = (lane < 16) ? 0 : 16;
#pragma unroll
    for (int nb = 0; nb < 4; ++nb) {
      FragU fb;
      const int ncol = nb * 16 + (lane & 15);
#pragma unroll
      for (int v = 0; v < 8; ++v)
        fb.u[v] = *(const unsigned int*)&Bs[ncol][kbb + 2 * v];
      acc[nb] = __builtin_amdgcn_wmma_f32_16x16x32_bf16(
          false, fa.v, false, fb.v, (short)0, acc[nb], false, false);
    }
    __syncthreads();
  }
  {
    const int ml = wave * 16 + ((lane >= 16) ? 8 : 0);
    const int nl = lane & 15;
#pragma unroll
    for (int nb = 0; nb < 4; ++nb)
#pragma unroll
      for (int r = 0; r < 8; ++r)
        Cs[ml + r][nb * 16 + nl] = acc[nb][r];
  }
  __syncthreads();
  for (int i = tid; i < 64 * 16; i += 128) {
    int r = i >> 4, c4 = i & 15;
    int m = bm0 + r;
    if (m >= M) continue;
    *(float4*)&Y[base + (size_t)m * str + c4 * 4] = *(float4*)&Cs[r][c4 * 4];
  }
}

// ---------------------------------------------------------------------------
// Masked softmax over last dim (in place); zero-fills [N, Npad).
// ---------------------------------------------------------------------------
__global__ __launch_bounds__(256) void attn_softmax(
    float* __restrict__ E, int mode, int M, int N, int Npad, int T,
    int An, int Gn, const int* __restrict__ pm,
    const int* __restrict__ mask2, const int* __restrict__ vmask)
{
  __shared__ float red[256];
  const int row = blockIdx.x;
  const int b = row / M, i = row - b * M;
  const int bt = b >> 2;
  const int tid = threadIdx.x;
  int rowvalid = (mode == 0) ? pm[bt * T + i] : pm[i * T + bt];
  float* rp = E + (size_t)row * Npad;
  float mx = NEGV;
  for (int j = tid; j < N; j += 256) {
    int valid = rowvalid;
    if (mode == 0)      valid = valid && pm[bt * T + j];
    else if (mode == 1) valid = valid && mask2[i * An + j] && pm[j * T + bt];
    else                valid = valid && mask2[i * Gn + j] && vmask[j * T + bt];
    float v = valid ? rp[j] : NEGV;
    rp[j] = v;
    mx = fmaxf(mx, v);
  }
  red[tid] = mx; __syncthreads();
  for (int s = 128; s > 0; s >>= 1) { if (tid < s) red[tid] = fmaxf(red[tid], red[tid + s]); __syncthreads(); }
  mx = red[0]; __syncthreads();
  float sum = 0.0f;
  for (int j = tid; j < N; j += 256) { float e = __expf(rp[j] - mx); rp[j] = e; sum += e; }
  red[tid] = sum; __syncthreads();
  for (int s = 128; s > 0; s >>= 1) { if (tid < s) red[tid] += red[tid + s]; __syncthreads(); }
  float inv = 1.0f / red[0];
  for (int j = tid; j < N; j += 256) rp[j] *= inv;
  for (int j = N + tid; j < Npad; j += 256) rp[j] = 0.0f;
}

// ---------------------------------------------------------------------------
// Small-K embedding (K=9/6): one block per row, 256 channels.
// ---------------------------------------------------------------------------
__global__ __launch_bounds__(256) void embed_linear(
    const float* __restrict__ X, const float* __restrict__ W,
    const float* __restrict__ bias, const int* __restrict__ rowmask,
    float* __restrict__ H, int M, int Kin)
{
  __shared__ float xr[16];
  const int r = blockIdx.x, c = threadIdx.x;
  if (c < Kin) {
    float v = X[(size_t)r * Kin + c];
    if (rowmask && !rowmask[r]) v = -1.0f;
    xr[c] = v;
  }
  __syncthreads();
  float acc = bias[c];
  for (int k = 0; k < Kin; ++k) acc += xr[k] * W[k * 256 + c];
  H[(size_t)r * 256 + c] = fmaxf(acc, 0.0f);
}

__global__ void zero_f32(float* p, int n) {
  int i = blockIdx.x * blockDim.x + threadIdx.x;
  if (i < n) p[i] = 0.0f;
}

__global__ __launch_bounds__(256) void bn_reduce(
    const float* __restrict__ H, float* __restrict__ sum,
    float* __restrict__ sqs, int M)
{
  const int c = threadIdx.x;
  float s = 0.0f, q = 0.0f;
  for (int r = blockIdx.x; r < M; r += gridDim.x) {
    float v = H[(size_t)r * 256 + c];
    s += v; q += v * v;
  }
  atomicAdd(&sum[c], s); atomicAdd(&sqs[c], q);
}

__global__ __launch_bounds__(256) void bn_apply(
    float* __restrict__ H, const float* __restrict__ sum,
    const float* __restrict__ sqs, const float* __restrict__ g,
    const float* __restrict__ beta, int M)
{
  const int c = threadIdx.x;
  float mu = sum[c] / (float)M;
  float var = sqs[c] / (float)M - mu * mu;
  float inv = rsqrtf(var + 1e-5f) * g[c];
  float bo = beta[c];
  for (int r = blockIdx.x; r < M; r += gridDim.x) {
    size_t idx = (size_t)r * 256 + c;
    H[idx] = (H[idx] - mu) * inv + bo;
  }
}

__global__ __launch_bounds__(256) void layernorm(
    const float* __restrict__ X, const float* __restrict__ g,
    const float* __restrict__ b, float* __restrict__ Y)
{
  __shared__ float red[256];
  const int r = blockIdx.x, c = threadIdx.x;
  float v = X[(size_t)r * 256 + c];
  red[c] = v; __syncthreads();
  for (int s = 128; s > 0; s >>= 1) { if (c < s) red[c] += red[c + s]; __syncthreads(); }
  float m = red[0] * (1.0f / 256.0f); __syncthreads();
  float d = v - m;
  red[c] = d * d; __syncthreads();
  for (int s = 128; s > 0; s >>= 1) { if (c < s) red[c] += red[c + s]; __syncthreads(); }
  float var = red[0] * (1.0f / 256.0f);
  Y[(size_t)r * 256 + c] = d * rsqrtf(var + 1e-5f) * g[c] + b[c];
}

// ---------------------------------------------------------------------------
extern "C" void kernel_launch(void* const* d_in, const int* in_sizes, int n_in,
                              void* d_out, int out_size, void* d_ws, size_t ws_size,
                              hipStream_t stream)
{
  (void)in_sizes; (void)n_in; (void)out_size; (void)ws_size;
  const float* state_feat = (const float*)d_in[0];
  const int*   bm_mask    = (const int*)d_in[1];
  const int*   pm         = (const int*)d_in[2];
  const int*   hm         = (const int*)d_in[3];
  const float* road_feat  = (const float*)d_in[4];
  const int*   rgv        = (const int*)d_in[5];
  const float* tl_feat    = (const float*)d_in[6];
  const int*   tlv        = (const int*)d_in[7];
  const int*   ar         = (const int*)d_in[8];
  const int*   atl        = (const int*)d_in[9];

  auto F = [&](int i) { return (const float*)d_in[i]; };
  const int EA = 10, EB = 14, EC = 18, ATT = 22;
  // att params: wk,bk,wv,bv,wq,bq,scale,wy,by,wf1,bf1,wf2,bf2,ln_g,ln_b
  auto AB = [&](int blk) { return ATT + blk * 15; };

  const int MA = A_NUM * T_LEN;   // 5824
  const int MC = G_RG * T_LEN;    // 127400
  const int MB = G_TL * T_LEN;    // 1456

  char* wsp = (char*)d_ws;
  auto alloc = [&](size_t nfloats) {
    float* p = (float*)wsp;
    wsp += ((nfloats * sizeof(float) + 255) / 256) * 256;
    return p;
  };
  float* xb  = alloc((size_t)MA * 256);
  float* tB  = alloc((size_t)MB * 256);
  float* tC  = alloc((size_t)MC * 256);
  float* Qb  = alloc((size_t)MA * 256);
  float* Kb  = alloc((size_t)MA * 256);
  float* Vb  = alloc((size_t)MA * 256);
  float* Kx  = alloc((size_t)MC * 256);
  float* Vx  = alloc((size_t)MC * 256);
  float* Eb  = alloc((size_t)(T_LEN * HEADS) * 64 * 1408);
  float* Y1  = alloc((size_t)MA * 256);
  float* Sb  = alloc((size_t)MA * 256);
  float* F1b = alloc((size_t)MA * 1024);
  float* F2b = alloc((size_t)MA * 256);
  float* bns = alloc(512);
  unsigned short* Wpack = (unsigned short*)alloc(256 * 1024 / 2);  // 262144 bf16

  auto gemm = [&](const float* Ap, const float* Wp, const float* bp,
                  const float* rp, const float* sp, float* Cp,
                  int M, int N, int K, int relu) {
    int tot = K * N;
    pack_wt<<<(tot + 255) / 256, 256, 0, stream>>>(Wp, Wpack, K, N);
    dim3 g((M + 63) / 64, N / 64);
    gemm_bf16<<<g, 128, 0, stream>>>(Ap, Wpack, bp, rp, sp, Cp, M, N, K, relu);
  };

  auto embed = [&](const float* X, int pb, const int* mask, float* Hout, int M, int Kin) {
    embed_linear<<<M, 256, 0, stream>>>(X, F(pb), F(pb + 1), mask, Hout, M, Kin);
    zero_f32<<<2, 256, 0, stream>>>(bns, 512);
    bn_reduce<<<256, 256, 0, stream>>>(Hout, bns, bns + 256, M);
    bn_apply<<<256, 256, 0, stream>>>(Hout, bns, bns + 256, F(pb + 2), F(pb + 3), M);
  };

  auto ffn = [&](int pb) {
    gemm(Y1,  F(pb + 7), F(pb + 8),  xb, nullptr, Sb,  MA, 256, 256, 1);
    gemm(Sb,  F(pb + 9), F(pb + 10), nullptr, nullptr, F1b, MA, 1024, 256, 1);
    gemm(F1b, F(pb + 11), F(pb + 12), nullptr, nullptr, F2b, MA, 256, 1024, 1);
    layernorm<<<MA, 256, 0, stream>>>(F2b, F(pb + 13), F(pb + 14), xb);
  };

  auto self_att = [&](int blk, bool across_time) {
    int pb = AB(blk);
    gemm(xb, F(pb + 0), F(pb + 1), nullptr, nullptr,  Kb, MA, 256, 256, 1);
    gemm(xb, F(pb + 2), F(pb + 3), nullptr, nullptr,  Vb, MA, 256, 256, 1);
    gemm(xb, F(pb + 4), F(pb + 5), nullptr, F(pb + 6), Qb, MA, 256, 256, 1);
    if (across_time) {
      const int B = A_NUM * HEADS, M = T_LEN, N = T_LEN, Np = 128;
      attn_qk<<<dim3(2, 2, B), 128, 0, stream>>>(Qb, Kb, Eb, 0, M, N, Np, T_LEN);
      attn_softmax<<<B * M, 256, 0, stream>>>(Eb, 0, M, N, Np, T_LEN, A_NUM, 0,
                                              pm, (const int*)nullptr, (const int*)nullptr);
      attn_av<<<dim3(2, 1, B), 128, 0, stream>>>(Eb, Vb, Y1, 0, M, N, Np, T_LEN);
    } else {
      const int B = T_LEN * HEADS, M = A_NUM, N = A_NUM, Np = 64;
      attn_qk<<<dim3(1, 1, B), 128, 0, stream>>>(Qb, Kb, Eb, 1, M, N, Np, T_LEN);
      attn_softmax<<<B * M, 256, 0, stream>>>(Eb, 1, M, N, Np, T_LEN, A_NUM, 0,
                                              pm, bm_mask, (const int*)nullptr);
      attn_av<<<dim3(1, 1, B), 128, 0, stream>>>(Eb, Vb, Y1, 1, M, N, Np, T_LEN);
    }
    ffn(pb);
  };

  auto cross_att = [&](int blk, const float* kvemb, int Gn,
                       const int* armask, const int* vmask) {
    int pb = AB(blk);
    int Mg = Gn * T_LEN;
    gemm(kvemb, F(pb + 0), F(pb + 1), nullptr, nullptr,  Kx, Mg, 256, 256, 1);
    gemm(kvemb, F(pb + 2), F(pb + 3), nullptr, nullptr,  Vx, Mg, 256, 256, 1);
    gemm(xb,    F(pb + 4), F(pb + 5), nullptr, F(pb + 6), Qb, MA, 256, 256, 1);
    const int B = T_LEN * HEADS, M = A_NUM, N = Gn;
    const int Np = ((Gn + 63) / 64) * 64;
    attn_qk<<<dim3(1, Np / 64, B), 128, 0, stream>>>(Qb, Kx, Eb, 2, M, N, Np, T_LEN);
    attn_softmax<<<B * M, 256, 0, stream>>>(Eb, 2, M, N, Np, T_LEN, A_NUM, Gn,
                                            pm, armask, vmask);
    attn_av<<<dim3(1, 1, B), 128, 0, stream>>>(Eb, Vx, Y1, 2, M, N, Np, T_LEN);
    ffn(pb);
  };

  embed(state_feat, EA, hm,      xb, MA, 9);
  embed(tl_feat,    EB, nullptr, tB, MB, 9);
  embed(road_feat,  EC, nullptr, tC, MC, 6);

  self_att(0, true);  self_att(1, false);
  self_att(2, true);  self_att(3, false);
  self_att(4, true);  self_att(5, false);
  cross_att(6, tC, G_RG, ar, rgv);
  cross_att(7, tB, G_TL, atl, tlv);
  self_att(8, true);  self_att(9, false);
  cross_att(10, tC, G_RG, ar, rgv);
  cross_att(11, tB, G_TL, atl, tlv);
  self_att(12, true); self_att(13, false);

  hipMemcpyAsync(d_out, xb, (size_t)MA * 256 * sizeof(float),
                 hipMemcpyDeviceToDevice, stream);
}
